// HyperGatedLinear_42167988912494
// MI455X (gfx1250) — matrix operand order repaired and checked by reading, not compiled, not run on hardware
//
#include <hip/hip_runtime.h>
#include <hip/hip_bf16.h>

// HyperGatedLinear on MI455X (gfx1250, wave32, WMMA bf16 path)
//
// out[b,o] = a*<base_w[o],x[b]> + (1-a)*( sum_k z[b,k]*T[b,o,k] + <dw_b[o],x[b]> )
//          + a*base_bias[o]     + (1-a)*( <z[b],db_W[o]> + db_b[o] )
// T[:, o, :] = x @ W_o,  W_o = dw_W[o*IN*ZD ..] viewed as [IN x ZD] (row-major, ZD contig)
//
// One block per o: GEMM  C[128 x 160] = x[128 x 1024] @ [W_o | base_w[o] | dw_b[o] | 0pad]
// Cols 0..127 -> z-contracted delta term; col 128 -> base dot; col 129 -> dw_b dot.

typedef __attribute__((ext_vector_type(16))) __bf16 v16bf;
typedef __attribute__((ext_vector_type(8)))  __bf16 v8bf;
typedef __attribute__((ext_vector_type(4)))  __bf16 v4bf;
typedef __attribute__((ext_vector_type(8)))  float  v8f;
typedef __attribute__((ext_vector_type(4)))  float  v4f;

#define B_    128
#define IN_   1024
#define OUT_  1024
#define ZD_   128
#define NB_   160   // padded GEMM-N: 128 z-cols + base col + dw_b col + 30 zero pad
#define QT    64    // K-dim (i) tile staged in LDS per iteration
#define LDA   72    // LDS row stride in bf16 elems (64 + 8 pad, keeps 16B alignment)
#define A_MIN 0.2f

__device__ __forceinline__ v4bf cvt4_bf16(v4f v) {
  v4bf r;
  r[0] = (__bf16)v[0]; r[1] = (__bf16)v[1];
  r[2] = (__bf16)v[2]; r[3] = (__bf16)v[3];
  return r;
}

__global__ __launch_bounds__(256)
void hyper_gated_linear_kernel(const float* __restrict__ x,
                               const float* __restrict__ z,
                               const float* __restrict__ base_weight,
                               const float* __restrict__ dw_W,
                               const float* __restrict__ dw_b,
                               const float* __restrict__ base_bias,
                               const float* __restrict__ db_W,
                               const float* __restrict__ db_b,
                               const float* __restrict__ s_ptr,
                               float* __restrict__ out)
{
  __shared__ __bf16 Alds[B_ * LDA];    // x tile       [m=0..127][q=0..63]
  __shared__ __bf16 Btld[NB_ * LDA];   // aug W tile^T [n=0..159][q=0..63]
  __shared__ float  sdelta[B_];
  __shared__ float  sbase[B_];
  __shared__ float  sdwb[B_];

  const int o    = blockIdx.x;
  const int t    = threadIdx.x;
  const int lane = t & 31;
  const int w    = t >> 5;      // wave 0..7
  const int hi   = lane >> 4;   // 0/1: lane half
  const int l16  = lane & 15;
  const int wm   = w & 3;       // owns m-tiles 2*wm, 2*wm+1      (m = b)
  const int wn   = w >> 2;      // owns n-tiles 5*wn .. 5*wn+4    (n = aug col)

  const float* __restrict__ wbase = dw_W + (size_t)o * (IN_ * ZD_);

  // zero the pad columns (rows 130..159) once; never rewritten inside the loop
  for (int i = t; i < (NB_ - ZD_ - 2) * LDA; i += 256)
    Btld[(ZD_ + 2) * LDA + i] = (__bf16)0.0f;

  v8f acc[2][5];
#pragma unroll
  for (int mi = 0; mi < 2; ++mi)
#pragma unroll
    for (int ni = 0; ni < 5; ++ni)
#pragma unroll
      for (int j = 0; j < 8; ++j) acc[mi][ni][j] = 0.0f;

  for (int q0 = 0; q0 < IN_; q0 += QT) {
    __syncthreads();  // previous tile fully consumed (also covers pad-zero init)

    // ---- stage A = x[:, q0:q0+64] as bf16, row-major [m][q] ----
#pragma unroll
    for (int r = 0; r < 8; ++r) {
      int f   = t + 256 * r;        // 0..2047 float4 slots
      int row = f >> 4;             // 0..127
      int qc  = (f & 15) << 2;      // 0..60
      v4f xv = *(const v4f*)(x + (size_t)row * IN_ + q0 + qc);
      *(v4bf*)(&Alds[row * LDA + qc]) = cvt4_bf16(xv);
    }

    // ---- stage B^T cols 0..127: 4q x 4n micro-blocks, b128 global reads (NT),
    //      transposed in registers, b64 LDS stores along contiguous q ----
#pragma unroll
    for (int r = 0; r < 2; ++r) {
      int s  = t + 256 * r;         // 0..511 micro-block slots
      int qb = (s >> 5) << 2;       // 0..60
      int nb = (s & 31) << 2;       // 0..124 (coalesced across a wave)
      const float* wp = wbase + (size_t)(q0 + qb) * ZD_ + nb;
      v4f w0 = __builtin_nontemporal_load((const v4f*)(wp + 0 * ZD_));
      v4f w1 = __builtin_nontemporal_load((const v4f*)(wp + 1 * ZD_));
      v4f w2 = __builtin_nontemporal_load((const v4f*)(wp + 2 * ZD_));
      v4f w3 = __builtin_nontemporal_load((const v4f*)(wp + 3 * ZD_));
#pragma unroll
      for (int k = 0; k < 4; ++k) {
        v4bf c;
        c[0] = (__bf16)w0[k]; c[1] = (__bf16)w1[k];
        c[2] = (__bf16)w2[k]; c[3] = (__bf16)w3[k];
        *(v4bf*)(&Btld[(nb + k) * LDA + qb]) = c;
      }
    }

    // ---- stage aug cols: n=128 -> base_weight[o], n=129 -> dw_b[o] ----
    if (t < 32) {
      int which = t >> 4;           // 0: base, 1: dw_b
      int q4    = (t & 15) << 2;    // 0..60
      const float* src = which ? (dw_b        + (size_t)o * IN_)
                               : (base_weight + (size_t)o * IN_);
      v4f v = *(const v4f*)(src + q0 + q4);
      *(v4bf*)(&Btld[(ZD_ + which) * LDA + q4]) = cvt4_bf16(v);
    }

    // prefetch next W tile (64 rows x 512 B): one 128 B line per thread
    if (q0 + QT < IN_)
      __builtin_prefetch(wbase + (size_t)(q0 + QT) * ZD_ + t * 32, 0, 3);

    __syncthreads();

    // ---- two K=32 WMMA steps over this 64-wide tile ----
#pragma unroll
    for (int ks = 0; ks < 2; ++ks) {
      const int qb = ks * 32;
      v16bf afrag[2], bfrag[5];
#pragma unroll
      for (int mi = 0; mi < 2; ++mi) {
        // A 16x32 bf16 frag: row = l16, elems 0..7 at q = qb+8*hi, 8..15 at +16
        int m = 16 * (2 * wm + mi) + l16;
        const __bf16* ap = &Alds[m * LDA + qb + 8 * hi];
        v8bf lo = *(const v8bf*)ap;
        v8bf hb = *(const v8bf*)(ap + 16);
#pragma unroll
        for (int e = 0; e < 8; ++e) { afrag[mi][e] = lo[e]; afrag[mi][8 + e] = hb[e]; }
      }
#pragma unroll
      for (int ni = 0; ni < 5; ++ni) {
        // B 32x16 bf16 frag: col = l16, elems 0..7 at q = qb+8*hi, 8..15 at +16
        int n = 16 * (5 * wn + ni) + l16;
        const __bf16* bp = &Btld[n * LDA + qb + 8 * hi];
        v8bf lo = *(const v8bf*)bp;
        v8bf hb = *(const v8bf*)(bp + 16);
#pragma unroll
        for (int e = 0; e < 8; ++e) { bfrag[ni][e] = lo[e]; bfrag[ni][8 + e] = hb[e]; }
      }
#pragma unroll
      for (int mi = 0; mi < 2; ++mi)
#pragma unroll
        for (int ni = 0; ni < 5; ++ni)
          acc[mi][ni] = __builtin_amdgcn_wmma_f32_16x16x32_bf16(
              false, afrag[mi], false, bfrag[ni],
              (short)0, acc[mi][ni], false, false);
    }
  }

  // ================= epilogue =================
  __syncthreads();
  if (t < B_) sdelta[t] = 0.0f;
  __syncthreads();

  // C/D layout: VGPR j holds row m = 8*hi + j (of the 16-tile), lane&15 = col.
  // z-weighted contraction of cols < 128; col 128 -> sbase, col 129 -> sdwb.
  const int nz_tiles = (wn == 0) ? 5 : 3;   // n-tiles fully inside ZD for this wave
#pragma unroll
  for (int mi = 0; mi < 2; ++mi) {
#pragma unroll
    for (int j = 0; j < 8; ++j) {
      int b = 16 * (2 * wm + mi) + 8 * hi + j;
      float pd = 0.0f;
#pragma unroll
      for (int ni = 0; ni < 5; ++ni) {
        if (ni < nz_tiles) {
          int n = 16 * (5 * wn + ni) + l16;
          pd += z[(size_t)b * ZD_ + n] * acc[mi][ni][j];
        }
      }
      atomicAdd(&sdelta[b], pd);            // ds_add_f32
      if (wn == 1) {                        // n-tile 8 holds cols 128..143
        if (l16 == 0) sbase[b] = acc[mi][3][j];   // exactly one producer per b
        if (l16 == 1) sdwb[b]  = acc[mi][3][j];   // exactly one producer per b
      }
    }
  }
  __syncthreads();

  if (t < B_) {
    const int b = t;
    const float* __restrict__ zr  = z    + (size_t)b * ZD_;
    const float* __restrict__ dbw = db_W + (size_t)o * ZD_;
    float zdb = 0.0f;
    for (int k = 0; k < ZD_; k += 4) {
      v4f zv = *(const v4f*)(zr + k);
      v4f dv = *(const v4f*)(dbw + k);
      zdb += zv[0] * dv[0] + zv[1] * dv[1] + zv[2] * dv[2] + zv[3] * dv[3];
    }
    const float s = s_ptr[0];
    const float a = A_MIN / (1.0f + expf(-s));  // a_min * sigmoid(s)
    const float res = a * (sbase[b] + base_bias[o])
                    + (1.0f - a) * (sdelta[b] + sdwb[b] + zdb + db_b[o]);
    out[(size_t)b * OUT_ + o] = res;
  }
}

extern "C" void kernel_launch(void* const* d_in, const int* in_sizes, int n_in,
                              void* d_out, int out_size, void* d_ws, size_t ws_size,
                              hipStream_t stream) {
  (void)in_sizes; (void)n_in; (void)out_size; (void)d_ws; (void)ws_size;
  const float* x           = (const float*)d_in[0];
  const float* z           = (const float*)d_in[1];
  const float* base_weight = (const float*)d_in[2];
  const float* dw_W        = (const float*)d_in[3];
  const float* dw_b        = (const float*)d_in[4];
  const float* base_bias   = (const float*)d_in[5];
  const float* db_W        = (const float*)d_in[6];
  const float* db_b        = (const float*)d_in[7];
  const float* s           = (const float*)d_in[8];
  float* out = (float*)d_out;

  hipLaunchKernelGGL(hyper_gated_linear_kernel, dim3(OUT_), dim3(256), 0, stream,
                     x, z, base_weight, dw_W, dw_b, base_bias, db_W, db_b, s, out);
}
